// FullCausalAttention_76424648065176
// MI455X (gfx1250) — compile-verified, matrix operand order checked by reading, not run on hardware
//
#include <hip/hip_runtime.h>
#include <hip/hip_bf16.h>

// ---------------------------------------------------------------------------
// Problem constants (from reference): B=2, N=2048, D=2048, H=16, HD=128
// ---------------------------------------------------------------------------
#define D_MODEL 2048
#define NH      16
#define HD      128
#define BATCH   2
#define SEQ     2048
#define MROWS   (BATCH * SEQ)   // 4096 rows of the "token" matrix

typedef __bf16 bf16_t;
typedef __attribute__((ext_vector_type(16))) __bf16 v16bf;
typedef __attribute__((ext_vector_type(8)))  __bf16 v8bf;
typedef __attribute__((ext_vector_type(4)))  __bf16 v4bf;
typedef __attribute__((ext_vector_type(8)))  float  v8f;
typedef __attribute__((ext_vector_type(4)))  float  v4f;

// WMMA fragment: 16 bf16 per lane = two 16-byte chunks
union Frag {
    v16bf v;
    v8bf  h[2];
};

static __device__ __forceinline__ v8f wmma_bf16(const Frag& a, const Frag& b, v8f c) {
    // D = A(16x32 bf16) x B(32x16 bf16) + C(16x16 f32)
    return __builtin_amdgcn_wmma_f32_16x16x32_bf16(false, a.v, false, b.v,
                                                   (short)0, c, false, false);
}

// Wave-scope LDS producer->consumer ordering (no workgroup barrier: waves in the
// attention kernel have divergent causal trip counts). DS ops of one wave are
// in-order; s_wait_dscnt 0 drains them, wave_barrier pins code motion.
static __device__ __forceinline__ void wave_lds_sync() {
    __builtin_amdgcn_wave_barrier();
    asm volatile("s_wait_dscnt 0" ::: "memory");
    __builtin_amdgcn_wave_barrier();
}

// CDNA5 async global->LDS copy (ASYNCcnt path, bypasses VGPR staging).
// lds_byte_addr: LDS-relative byte address (= low 32 bits of generic pointer).
static __device__ __forceinline__ void async_load_b128(unsigned lds_byte_addr,
                                                       const void* gaddr) {
    asm volatile("global_load_async_to_lds_b128 %0, %1, off"
                 :: "v"(lds_byte_addr), "v"(gaddr)
                 : "memory");
}
static __device__ __forceinline__ unsigned lds_addr(const void* p) {
    return (unsigned)(size_t)p;   // generic LDS address: addr[31:0] = LDS offset
}

// ---------------------------------------------------------------------------
// f32 -> bf16 flat conversion
// ---------------------------------------------------------------------------
__global__ __launch_bounds__(256)
void cvt_f32_bf16(const float* __restrict__ src, bf16_t* __restrict__ dst, int n) {
    int i = (blockIdx.x * 256 + threadIdx.x) * 4;
    if (i < n) {
        v4f f = *(const v4f*)(src + i);
        v4bf o;
#pragma unroll
        for (int e = 0; e < 4; ++e) o[e] = (bf16_t)f[e];
        *(v4bf*)(dst + i) = o;
    }
}

// ---------------------------------------------------------------------------
// W[R,C] f32  ->  WT[C,R] bf16  (tiled transpose through LDS)
// ---------------------------------------------------------------------------
__global__ __launch_bounds__(256)
void transpose_cvt(const float* __restrict__ W, bf16_t* __restrict__ WT, int R, int C) {
    __shared__ float tile[32][33];
    const int tx = threadIdx.x & 31;
    const int ty = threadIdx.x >> 5;          // 0..7
    const int c0 = blockIdx.x * 32;
    const int r0 = blockIdx.y * 32;
#pragma unroll
    for (int i = 0; i < 32; i += 8)
        tile[ty + i][tx] = W[(size_t)(r0 + ty + i) * C + (c0 + tx)];
    __syncthreads();
#pragma unroll
    for (int i = 0; i < 32; i += 8)
        WT[(size_t)(c0 + ty + i) * R + (r0 + tx)] = (bf16_t)tile[tx][ty + i];
}

// ---------------------------------------------------------------------------
// bf16 WMMA GEMM:  C[m,n] = sum_k A[m,k] * BT[n,k]  (+bias, templated epilogue)
//   Block tile 128x128, 8 waves (2x4), wave tile 64x32, K-step 32.
//   Double-buffered LDS fed by GLOBAL_LOAD_ASYNC_TO_LDS_B128: tile i+1 streams
//   in over the async path while tile i is consumed by WMMAs; s_wait_asynccnt 4
//   drains only the older tile (async loads complete in order).
// MODE 0: QKV epilogue  -> q[B,H,N,HD], k[B,H,N,HD], vT[B,H,HD,N] (bf16)
// MODE 1: gate epilogue -> o1 = sigmoid(acc+bias) * aux   (bf16)
// MODE 2: final         -> outf = acc + bias              (f32)
// ---------------------------------------------------------------------------
template<int MODE>
__global__ __launch_bounds__(256)
void gemm_wmma(const bf16_t* __restrict__ A, const bf16_t* __restrict__ BT,
               const float* __restrict__ bias, int Nc, int K,
               float* __restrict__ outf,
               bf16_t* __restrict__ o1, bf16_t* __restrict__ o2, bf16_t* __restrict__ o3,
               const bf16_t* __restrict__ aux)
{
    __shared__ bf16_t sA[2][128 * 32];   // [row][k] 64B rows
    __shared__ bf16_t sB[2][128 * 32];   // [col][k] (BT rows)

    const int tid  = threadIdx.x;
    const int lane = tid & 31;
    const int wave = tid >> 5;
    const int wm   = wave & 1;        // 0..1 -> 64-row half
    const int wn   = wave >> 1;       // 0..3 -> 32-col quarter
    const size_t m0 = (size_t)blockIdx.y * 128;
    const size_t n0 = (size_t)blockIdx.x * 128;

    const int r  = lane & 15;
    const int kb = (lane >> 4) * 8;
    const int rg = (lane >> 4) * 8;

    // this thread's two 16B chunks of each 128x32 tile (512 chunks total)
    const int rA0 = tid >> 2,          kcA0 = (tid & 3) * 8;
    const int rA1 = (tid + 256) >> 2,  kcA1 = ((tid + 256) & 3) * 8;

    v8f acc[4][2];
#pragma unroll
    for (int i = 0; i < 4; ++i)
#pragma unroll
        for (int j = 0; j < 2; ++j)
#pragma unroll
            for (int e = 0; e < 8; ++e) acc[i][j][e] = 0.f;

    auto issue_tile = [&](int buf, int k0) {
        async_load_b128(lds_addr(&sA[buf][rA0 * 32 + kcA0]), A  + (m0 + rA0) * K + k0 + kcA0);
        async_load_b128(lds_addr(&sA[buf][rA1 * 32 + kcA1]), A  + (m0 + rA1) * K + k0 + kcA1);
        async_load_b128(lds_addr(&sB[buf][rA0 * 32 + kcA0]), BT + (n0 + rA0) * K + k0 + kcA0);
        async_load_b128(lds_addr(&sB[buf][rA1 * 32 + kcA1]), BT + (n0 + rA1) * K + k0 + kcA1);
    };

    issue_tile(0, 0);
    int buf = 0;
    for (int k0 = 0; k0 < K; k0 += 32) {
        const bool more = (k0 + 32 < K);
        if (more) {
            issue_tile(buf ^ 1, k0 + 32);                       // 8 outstanding
            asm volatile("s_wait_asynccnt 4" ::: "memory");     // oldest tile landed
        } else {
            asm volatile("s_wait_asynccnt 0" ::: "memory");
        }
        __syncthreads();   // all waves' chunks of tile(buf) visible

        Frag bfrag[2];
#pragma unroll
        for (int j = 0; j < 2; ++j) {
            const bf16_t* p = &sB[buf][(wn * 32 + j * 16 + r) * 32];
            bfrag[j].h[0] = *(const v8bf*)(p + kb);
            bfrag[j].h[1] = *(const v8bf*)(p + 16 + kb);
        }
#pragma unroll
        for (int i = 0; i < 4; ++i) {
            Frag afrag;
            const bf16_t* p = &sA[buf][(wm * 64 + i * 16 + r) * 32];
            afrag.h[0] = *(const v8bf*)(p + kb);
            afrag.h[1] = *(const v8bf*)(p + 16 + kb);
#pragma unroll
            for (int j = 0; j < 2; ++j)
                acc[i][j] = wmma_bf16(afrag, bfrag[j], acc[i][j]);
        }
        __syncthreads();   // reads of tile(buf) done before it is re-targeted
        buf ^= 1;
    }

    // Epilogue. C-layout: lane -> col = (lane&15); VGPR e -> row = (lane>>4)*8+e
#pragma unroll
    for (int i = 0; i < 4; ++i) {
#pragma unroll
        for (int j = 0; j < 2; ++j) {
            const int gn   = (int)n0 + wn * 32 + j * 16 + r;
            const float bv = bias[gn];
#pragma unroll
            for (int e = 0; e < 8; ++e) {
                const int gm = (int)m0 + wm * 64 + i * 16 + rg + e;
                float v = acc[i][j][e] + bv;
                if (MODE == 0) {
                    const int which = gn >> 11;        // 0=q 1=k 2=v (uniform/block)
                    const int dcol  = gn & 2047;
                    const int h  = dcol >> 7, hd = dcol & 127;
                    const int b  = gm >> 11, nr = gm & 2047;
                    const size_t bh = (size_t)(b * NH + h);
                    if (which == 0)      o1[(bh * SEQ + nr) * HD + hd] = (bf16_t)v;
                    else if (which == 1) o2[(bh * SEQ + nr) * HD + hd] = (bf16_t)v;
                    else                 o3[(bh * HD + hd) * SEQ + nr] = (bf16_t)v; // V^T
                } else if (MODE == 1) {
                    const float g = 1.f / (1.f + __expf(-v));
                    const float o = (float)aux[(size_t)gm * D_MODEL + gn];
                    o1[(size_t)gm * D_MODEL + gn] = (bf16_t)(g * o);
                } else {
                    outf[(size_t)gm * D_MODEL + gn] = v;
                }
            }
        }
    }
}

// ---------------------------------------------------------------------------
// Causal flash attention. Grid: (SEQ/64, B*H), block = 4 waves.
// Each wave owns 16 query rows; KV blocks of 32; online softmax via per-wave
// LDS tiles (S f32, P bf16) with wave-scope DS sync only.
//   Q,K: [B,H,N,HD] bf16     V^T: [B,H,HD,N] bf16     O: [B,N,D] bf16
// ---------------------------------------------------------------------------
__global__ __launch_bounds__(128)
void attn_flash(const bf16_t* __restrict__ Q, const bf16_t* __restrict__ Km,
                const bf16_t* __restrict__ Vt, bf16_t* __restrict__ O)
{
    const int lane = threadIdx.x & 31;
    const int wave = threadIdx.x >> 5;
    const int bh   = blockIdx.y;                    // b*NH + h
    const int q0   = blockIdx.x * 64 + wave * 16;   // this wave's first query row
    const float sc = 0.08838834764831845f;          // 1/sqrt(128)

    const bf16_t* Qh = Q  + (size_t)bh * SEQ * HD;
    const bf16_t* Kh = Km + (size_t)bh * SEQ * HD;
    const bf16_t* Vh = Vt + (size_t)bh * HD * SEQ;

    __shared__ float  sS[4][16 * 32];
    __shared__ bf16_t sP[4][16 * 32];
    __shared__ float  sM[4][16], sL[4][16], sC[4][16];
    float*  S  = sS[wave];
    bf16_t* P  = sP[wave];
    float*  Mr = sM[wave];
    float*  Lr = sL[wave];
    float*  Cr = sC[wave];

    const int r  = lane & 15;
    const int kb = (lane >> 4) * 8;
    const int rg = (lane >> 4) * 8;

    // Q fragments for 4 K-steps of 32 (held for the whole KV loop)
    Frag qf[4];
    {
        const bf16_t* qrow = Qh + (size_t)(q0 + r) * HD;
#pragma unroll
        for (int s = 0; s < 4; ++s) {
            qf[s].h[0] = *(const v8bf*)(qrow + s * 32 + kb);
            qf[s].h[1] = *(const v8bf*)(qrow + s * 32 + 16 + kb);
        }
    }

    v8f oacc[8];
#pragma unroll
    for (int t = 0; t < 8; ++t)
#pragma unroll
        for (int e = 0; e < 8; ++e) oacc[t][e] = 0.f;

    if (lane < 16) { Mr[lane] = -3.0e38f; Lr[lane] = 0.f; }
    wave_lds_sync();

    for (int kv0 = 0; kv0 <= q0 + 15; kv0 += 32) {
        // ---- S = Q K^T for 16x32 block (2 col-tiles x 4 K-steps) ----
#pragma unroll
        for (int ct = 0; ct < 2; ++ct) {
            v8f sacc;
#pragma unroll
            for (int e = 0; e < 8; ++e) sacc[e] = 0.f;
            const bf16_t* krow = Kh + (size_t)(kv0 + ct * 16 + r) * HD;
#pragma unroll
            for (int s = 0; s < 4; ++s) {
                Frag kf;
                kf.h[0] = *(const v8bf*)(krow + s * 32 + kb);
                kf.h[1] = *(const v8bf*)(krow + s * 32 + 16 + kb);
                sacc = wmma_bf16(qf[s], kf, sacc);
            }
            const int cc = ct * 16 + r;
#pragma unroll
            for (int e = 0; e < 8; ++e) S[(rg + e) * 32 + cc] = sacc[e];
        }
        wave_lds_sync();

        // ---- online softmax: lane L (<16) owns full row L ----
        if (lane < 16) {
            const int gq = q0 + lane;
            float mx = Mr[lane];
#pragma unroll
            for (int c = 0; c < 32; ++c) {
                float s = S[lane * 32 + c] * sc;
                s = (kv0 + c > gq) ? -3.0e38f : s;
                mx = fmaxf(mx, s);
            }
            const float corr = __expf(Mr[lane] - mx);
            float l = Lr[lane] * corr;
#pragma unroll
            for (int c = 0; c < 32; ++c) {
                float s = S[lane * 32 + c] * sc;
                float p = (kv0 + c > gq) ? 0.f : __expf(s - mx);
                l += p;
                P[lane * 32 + c] = (bf16_t)p;
            }
            Mr[lane] = mx; Lr[lane] = l; Cr[lane] = corr;
        }
        wave_lds_sync();

        // ---- O = corr*O + P V : P in A-layout from LDS, V^T rows from global ----
        Frag pf;
        pf.h[0] = *(const v8bf*)(P + r * 32 + kb);
        pf.h[1] = *(const v8bf*)(P + r * 32 + 16 + kb);
        float cr[8];
#pragma unroll
        for (int e = 0; e < 8; ++e) cr[e] = Cr[rg + e];

#pragma unroll
        for (int t = 0; t < 8; ++t) {
#pragma unroll
            for (int e = 0; e < 8; ++e) oacc[t][e] *= cr[e];
            const bf16_t* vrow = Vh + (size_t)(t * 16 + r) * SEQ + kv0;
            Frag vf;
            vf.h[0] = *(const v8bf*)(vrow + kb);
            vf.h[1] = *(const v8bf*)(vrow + 16 + kb);
            oacc[t] = wmma_bf16(pf, vf, oacc[t]);
        }
        wave_lds_sync();
    }

    // ---- normalize and store to [B,N,D] ----
    float invl[8];
#pragma unroll
    for (int e = 0; e < 8; ++e) invl[e] = 1.f / Lr[rg + e];

    const int b = bh >> 4, h = bh & 15;
#pragma unroll
    for (int t = 0; t < 8; ++t) {
#pragma unroll
        for (int e = 0; e < 8; ++e) {
            const int row = rg + e;
            const int hd  = t * 16 + r;
            O[((size_t)b * SEQ + (q0 + row)) * D_MODEL + h * HD + hd] =
                (bf16_t)(oacc[t][e] * invl[e]);
        }
    }
}

// ---------------------------------------------------------------------------
// Launcher
// ---------------------------------------------------------------------------
extern "C" void kernel_launch(void* const* d_in, const int* in_sizes, int n_in,
                              void* d_out, int out_size, void* d_ws, size_t ws_size,
                              hipStream_t stream)
{
    const float* x    = (const float*)d_in[0];
    const float* Wqkv = (const float*)d_in[1];
    const float* bqkv = (const float*)d_in[2];
    const float* Wo   = (const float*)d_in[3];
    const float* bo   = (const float*)d_in[4];
    const float* Wg   = (const float*)d_in[5];
    const float* bg   = (const float*)d_in[6];

    char* p = (char*)d_ws;
    auto carve = [&](size_t bytes) {
        char* r = p;
        p += (bytes + 255) & ~(size_t)255;
        return r;
    };
    bf16_t* xbf   = (bf16_t*)carve((size_t)MROWS * D_MODEL * 2);        // x bf16
    bf16_t* WqkvT = (bf16_t*)carve((size_t)3 * D_MODEL * D_MODEL * 2);  // [3D, D]
    bf16_t* WgT   = (bf16_t*)carve((size_t)D_MODEL * D_MODEL * 2);
    bf16_t* WoT   = (bf16_t*)carve((size_t)D_MODEL * D_MODEL * 2);
    bf16_t* qb    = (bf16_t*)carve((size_t)MROWS * D_MODEL * 2);        // [B,H,N,HD]
    bf16_t* kbuf  = (bf16_t*)carve((size_t)MROWS * D_MODEL * 2);        // [B,H,N,HD]
    bf16_t* vT    = (bf16_t*)carve((size_t)MROWS * D_MODEL * 2);        // [B,H,HD,N]
    bf16_t* ob    = (bf16_t*)carve((size_t)MROWS * D_MODEL * 2);        // attn out [B,N,D]
    bf16_t* gb    = (bf16_t*)carve((size_t)MROWS * D_MODEL * 2);        // gated out

    // 1) precision conversion + weight transposes (bandwidth-trivial vs compute)
    cvt_f32_bf16<<<(MROWS * D_MODEL) / (256 * 4), 256, 0, stream>>>(x, xbf, MROWS * D_MODEL);
    transpose_cvt<<<dim3(3 * D_MODEL / 32, D_MODEL / 32), 256, 0, stream>>>(Wqkv, WqkvT, D_MODEL, 3 * D_MODEL);
    transpose_cvt<<<dim3(D_MODEL / 32, D_MODEL / 32), 256, 0, stream>>>(Wg, WgT, D_MODEL, D_MODEL);
    transpose_cvt<<<dim3(D_MODEL / 32, D_MODEL / 32), 256, 0, stream>>>(Wo, WoT, D_MODEL, D_MODEL);

    // 2) QKV projection (writes q,k,vT in attention-friendly layouts)
    gemm_wmma<0><<<dim3(3 * D_MODEL / 128, MROWS / 128), 256, 0, stream>>>(
        xbf, WqkvT, bqkv, 3 * D_MODEL, D_MODEL, nullptr, qb, kbuf, vT, nullptr);

    // 3) causal flash attention
    attn_flash<<<dim3(SEQ / 64, BATCH * NH), 128, 0, stream>>>(qb, kbuf, vT, ob);

    // 4) gate GEMM fused with sigmoid * attn-out
    gemm_wmma<1><<<dim3(D_MODEL / 128, MROWS / 128), 256, 0, stream>>>(
        xbf, WgT, bg, D_MODEL, D_MODEL, nullptr, gb, nullptr, nullptr, ob);

    // 5) output projection -> f32 d_out
    gemm_wmma<2><<<dim3(D_MODEL / 128, MROWS / 128), 256, 0, stream>>>(
        gb, WoT, bo, D_MODEL, D_MODEL, (float*)d_out, nullptr, nullptr, nullptr, nullptr);
}
